// DiffAttention_68891275427847
// MI455X (gfx1250) — matrix operand order repaired and checked by reading, not compiled, not run on hardware
//
#include <hip/hip_runtime.h>

typedef __attribute__((ext_vector_type(16))) _Float16 v16h;
typedef __attribute__((ext_vector_type(8)))  _Float16 v8h;
typedef __attribute__((ext_vector_type(2)))  __fp16   h2;    // builtin half2 type
typedef __attribute__((ext_vector_type(8)))  float    v8f;
typedef __attribute__((ext_vector_type(4)))  float    v4f;

#define NN 4096
#define LL 4096
#define HH 8
#define MM 64
#define DD 64
#define STRIDE (HH*MM)          // 512 floats between consecutive n/l rows
#define ROWS_PER_BLOCK 128      // 8 waves x 16 rows
#define CHUNK 64                // keys/values per LDS buffer (one barrier per 64)
#define NCHUNK (LL/CHUNK)       // 64

union FragU { v16h f; v8h h[2]; unsigned u[8]; };
union H2U   { h2 v; _Float16 e[2]; unsigned u; };

__device__ __forceinline__ unsigned pk16(float a, float b) {
  // v_cvt_pk_rtz_f16_f32: two f32 -> packed f16 in one VALU op
  H2U r; r.v = __builtin_amdgcn_cvt_pkrtz(a, b); return r.u;
}

__device__ __forceinline__ float sigf(float x) {
#if __has_builtin(__builtin_amdgcn_tanhf)
  // sigmoid(x) = 0.5*tanh(x/2) + 0.5   (mul + v_tanh_f32 + fma)
  return __builtin_fmaf(0.5f, __builtin_amdgcn_tanhf(x * 0.5f), 0.5f);
#else
  // sigmoid(x) = 1 / (1 + 2^(-x*log2 e))  (mul + v_exp_f32 + add + v_rcp_f32)
  float e = __builtin_amdgcn_exp2f(x * -1.44269504088896340736f);
  return __builtin_amdgcn_rcpf(1.0f + e);
#endif
}

__global__ __launch_bounds__(256) void sig_attn_kernel(
    const float* __restrict__ q, const float* __restrict__ k,
    const float* __restrict__ v, float* __restrict__ out)
{
  __shared__ _Float16 ldsK[2][CHUNK * MM];   // [l][m] row-major f16   (16KB)
  __shared__ _Float16 ldsV[2][DD * CHUNK];   // [d][l] transposed f16  (16KB)

  const int h    = blockIdx.y;
  const int nb   = blockIdx.x * ROWS_PER_BLOCK;
  const int tid  = threadIdx.x;
  const int lane = tid & 31;
  const int wave = tid >> 5;
  const int ln16 = lane & 15;
  const int hi   = (lane >> 4) & 1;
  const int kb   = hi * 16;                  // A-fragment K base

  // ---- Q fragments: B operand of S^T = K*Q^T. col n = ln16, K-dim = m.
  const float* qrow = q + (size_t)(nb + wave*16 + ln16) * STRIDE + h * MM;
  FragU qf[2];
  #pragma unroll
  for (int s = 0; s < 2; ++s) {
    int mb = s*32 + hi*16;
    #pragma unroll
    for (int j = 0; j < 8; ++j) {
      H2U p; p.u = pk16(qrow[mb + 2*j], qrow[mb + 2*j + 1]);
      qf[s].f[2*j] = p.e[0]; qf[s].f[2*j + 1] = p.e[1];
    }
  }

  // ---- cooperative loader: 64 K rows (row-major) + 64 V rows (transposed)
  auto load_chunk = [&](int c, int b) {
    int l  = tid >> 2;          // 0..63
    int e0 = (tid & 3) << 4;    // 0,16,32,48
    const float* kp = k + (size_t)(c*CHUNK + l) * STRIDE + h * MM + e0;
    v8h k0, k1;
    #pragma unroll
    for (int j = 0; j < 4; ++j) {
      H2U p; p.u = pk16(kp[2*j],     kp[2*j + 1]);     k0[2*j] = p.e[0]; k0[2*j+1] = p.e[1];
      H2U r; r.u = pk16(kp[8 + 2*j], kp[8 + 2*j + 1]); k1[2*j] = r.e[0]; k1[2*j+1] = r.e[1];
    }
    *(v8h*)&ldsK[b][l * MM + e0]     = k0;
    *(v8h*)&ldsK[b][l * MM + e0 + 8] = k1;
    const float* vp = v + (size_t)(c*CHUNK + l) * STRIDE + h * DD + e0;
    #pragma unroll
    for (int j = 0; j < 8; ++j) {
      H2U p; p.u = pk16(vp[2*j], vp[2*j + 1]);
      ldsV[b][(e0 + 2*j)     * CHUNK + l] = p.e[0];
      ldsV[b][(e0 + 2*j + 1) * CHUNK + l] = p.e[1];
    }
  };

  v8f acc[4] = {};        // out^T accumulators: 64 d-rows x 16 n-cols
  float zacc = 0.0f;      // per-lane partial normalizer (fixed n = ln16)

#if __has_builtin(__builtin_amdgcn_fdot2)
  h2 ones = { (__fp16)1.0f, (__fp16)1.0f };
#endif

  load_chunk(0, 0);
  __syncthreads();

  for (int c = 0; c < NCHUNK; ++c) {
    if (c + 1 < NCHUNK) load_chunk(c + 1, (c + 1) & 1);

    const _Float16* Kb = ldsK[c & 1];
    const _Float16* Vb = ldsV[c & 1];

    #pragma unroll
    for (int s2 = 0; s2 < 2; ++s2) {         // two 32-key sub-tiles per buffer
      const int lb = s2 * 32;

      // ---- S^T tiles: D = K_rows x Q^T, two 16-row l-tiles, K=64 in 2 steps
      v8f s0 = {}, s1 = {};
      #pragma unroll
      for (int s = 0; s < 2; ++s) {
        FragU ka0, ka1;
        int b0 = (lb + ln16)      * MM + s*32 + kb;
        int b1 = (lb + 16 + ln16) * MM + s*32 + kb;
        ka0.h[0] = *(const v8h*)&Kb[b0];   ka0.h[1] = *(const v8h*)&Kb[b0 + 8];
        ka1.h[0] = *(const v8h*)&Kb[b1];   ka1.h[1] = *(const v8h*)&Kb[b1 + 8];
        s0 = __builtin_amdgcn_wmma_f32_16x16x32_f16(false, ka0.f, false, qf[s].f,
                                                    (short)0, s0, false, false);
        s1 = __builtin_amdgcn_wmma_f32_16x16x32_f16(false, ka1.f, false, qf[s].f,
                                                    (short)0, s1, false, false);
      }

      // ---- sigmoid, pack to f16 pairs, accumulate Z from packed halves
      unsigned t0p[4], t1p[4];
      #pragma unroll
      for (int j = 0; j < 4; ++j) {
        float a0 = sigf(s0[2*j]), a1 = sigf(s0[2*j + 1]);
        float b0 = sigf(s1[2*j]), b1 = sigf(s1[2*j + 1]);
        t0p[j] = pk16(a0, a1);
        t1p[j] = pk16(b0, b1);
#if __has_builtin(__builtin_amdgcn_fdot2)
        H2U x; x.u = t0p[j];
        H2U y; y.u = t1p[j];
        zacc = __builtin_amdgcn_fdot2(x.v, ones, zacc, false);
        zacc = __builtin_amdgcn_fdot2(y.v, ones, zacc, false);
#else
        zacc += (a0 + a1) + (b0 + b1);
#endif
      }

      // ---- assemble P^T B fragment: half-wave swap only
      FragU pf;
      #pragma unroll
      for (int j = 0; j < 4; ++j) {
        unsigned x0 = (unsigned)__shfl_xor((int)t0p[j], 16, 32);
        unsigned x1 = (unsigned)__shfl_xor((int)t1p[j], 16, 32);
        pf.u[j]     = hi ? x1     : t0p[j];
        pf.u[4 + j] = hi ? t1p[j] : x0;
      }

      // ---- out^T += V^T x P^T  (4 d-tiles)
      #pragma unroll
      for (int t = 0; t < 4; ++t) {
        FragU vf;
        int db = (t*16 + ln16) * CHUNK + lb + kb;
        vf.h[0] = *(const v8h*)&Vb[db];   vf.h[1] = *(const v8h*)&Vb[db + 8];
        acc[t] = __builtin_amdgcn_wmma_f32_16x16x32_f16(false, vf.f, false, pf.f,
                                                        (short)0, acc[t], false, false);
      }
    }

    __syncthreads();
  }

  // ---- close Z (pair lane n with lane n+16), divide once, store
  float ztot = zacc + __shfl_xor(zacc, 16, 32);
  float rz = 1.0f / ztot;

  float* orow = out + (size_t)(nb + wave*16 + ln16) * STRIDE + h * DD;
  #pragma unroll
  for (int t = 0; t < 4; ++t) {
    v4f lo, hv;
    #pragma unroll
    for (int r = 0; r < 4; ++r) { lo[r] = acc[t][r] * rz; hv[r] = acc[t][4 + r] * rz; }
    float* p = orow + t*16 + hi*8;       // element r -> d = t*16 + 8*hi + r
    *(v4f*)p       = lo;
    *(v4f*)(p + 4) = hv;
  }
}

extern "C" void kernel_launch(void* const* d_in, const int* in_sizes, int n_in,
                              void* d_out, int out_size, void* d_ws, size_t ws_size,
                              hipStream_t stream) {
  const float* q = (const float*)d_in[0];   // [N,H,M]
  const float* k = (const float*)d_in[1];   // [L,H,M]
  const float* v = (const float*)d_in[2];   // [L,H,D]
  float* out = (float*)d_out;               // [N,H,D]
  dim3 grid(NN / ROWS_PER_BLOCK, HH);
  sig_attn_kernel<<<grid, 256, 0, stream>>>(q, k, v, out);
}
